// GMVAE_78039555768659
// MI455X (gfx1250) — matrix operand order repaired and checked by reading, not compiled, not run on hardware
//
#include <hip/hip_runtime.h>

// Problem constants (match reference setup_inputs)
constexpr int cB   = 16;
constexpr int cN   = 1024;
constexpr int cHID = 256;
constexpr int cL   = 256;
constexpr int cMC  = 10;
constexpr int cK   = 64;            // GMM clusters
constexpr int cM   = cN * cMC;      // 10240 samples per batch
constexpr int cBN  = cB * cN;       // 16384 rows for MLP
constexpr int EM_ITERS = 5;
#define LOG_NORM (-0.91893853320467274178f)  // -0.5*ln(2*pi)

typedef __attribute__((ext_vector_type(16))) __bf16 v16bf;
typedef __attribute__((ext_vector_type(8)))  float  v8f;

// ---------------- wave32 reductions ----------------
__device__ __forceinline__ float wave_sum(float v) {
#pragma unroll
  for (int off = 16; off > 0; off >>= 1) v += __shfl_xor(v, off, 32);
  return v;
}
__device__ __forceinline__ float wave_max(float v) {
#pragma unroll
  for (int off = 16; off > 0; off >>= 1) v = fmaxf(v, __shfl_xor(v, off, 32));
  return v;
}

__device__ __forceinline__ v8f wmma_bf16(v16bf a, v16bf b, v8f c) {
  return __builtin_amdgcn_wmma_f32_16x16x32_bf16(false, a, false, b, (short)0,
                                                 c, false, false);
}

__device__ __forceinline__ void wait_asynccnt0() {
#if __has_builtin(__builtin_amdgcn_s_wait_asynccnt)
  __builtin_amdgcn_s_wait_asynccnt(0);
#else
  asm volatile("s_wait_asynccnt 0" ::: "memory");
#endif
}

// ---------------- WMMA bf16 fragment loaders (ISA 7.12.2 16-bit layouts) ----
// A (16x32, MxK): lane l: m = l&15; VGPR v holds K pair
//   k = (v>>2)*16 + (l>>4)*8 + (v&3)*2
template <typename T>
__device__ __forceinline__ v16bf load_frag_a(const T* __restrict__ A, int lda,
                                             int lane, int kk) {
  const int row = lane & 15, hi = lane >> 4;
  v16bf f;
#pragma unroll
  for (int v = 0; v < 8; ++v) {
    const int k = kk + ((v >> 2) << 4) + (hi << 3) + ((v & 3) << 1);
    const T* p = A + (size_t)row * lda + k;
    f[2 * v]     = (__bf16)p[0];
    f[2 * v + 1] = (__bf16)p[1];
  }
  return f;
}
// B (32x16, KxN) loaded from row-major B^T (NxK): lane l: n = l&15;
//   VGPR v holds K pair k = (v>>2)*16 + (l>>4)*8 + (v&3)*2   (same as A)
template <typename T>
__device__ __forceinline__ v16bf load_frag_b(const T* __restrict__ Bt, int ldb,
                                             int lane, int kk) {
  const int col = lane & 15, hi = lane >> 4;
  v16bf f;
#pragma unroll
  for (int v = 0; v < 8; ++v) {
    const int k = kk + ((v >> 2) << 4) + (hi << 3) + ((v & 3) << 1);
    const T* p = Bt + (size_t)col * ldb + k;
    f[2 * v]     = (__bf16)p[0];
    f[2 * v + 1] = (__bf16)p[1];
  }
  return f;
}

// ---------------- generic batched wave-tile WMMA GEMM ----------------------
// C[b] (MxN, f32 accum) = A[b] (MxK) * Bt[b]^T  (Bt is NxK row-major, bf16)
// Each wave computes a 32x32 tile (2x2 WMMA tiles): 4 WMMAs per 4 frag loads.
// EPI: 0 = none, 1 = sigmoid, 2 = divide row m by rowScale[b][m]
template <typename TA, typename TO, int EPI, bool HAS_BIAS>
__global__ void wmma_gemm_kernel(const TA* __restrict__ A, int lda, size_t sA,
                                 const __bf16* __restrict__ Bt, int ldb, size_t sB,
                                 TO* __restrict__ C, int ldc, size_t sC,
                                 const float* __restrict__ bias,
                                 const float* __restrict__ rowScale, int sScale,
                                 int M, int N, int K) {
  const int wave = threadIdx.x >> 5;
  const int lane = threadIdx.x & 31;
  const int tn2 = N >> 5;
  const int tm2 = M >> 5;
  const int tile = blockIdx.x * (blockDim.x >> 5) + wave;
  if (tile >= tm2 * tn2) return;            // wave-uniform guard
  const int tm = tile / tn2, tn = tile % tn2;
  const int b = blockIdx.y;

  const TA*     Ab = A  + (size_t)b * sA + (size_t)tm * 32 * lda;
  const __bf16* Bb = Bt + (size_t)b * sB + (size_t)tn * 32 * ldb;

  v8f acc[2][2] = {};
  for (int kk = 0; kk < K; kk += 32) {
    v16bf a0 = load_frag_a(Ab,                    lda, lane, kk);
    v16bf a1 = load_frag_a(Ab + (size_t)16 * lda, lda, lane, kk);
    v16bf b0 = load_frag_b(Bb,                    ldb, lane, kk);
    v16bf b1 = load_frag_b(Bb + (size_t)16 * ldb, ldb, lane, kk);
    acc[0][0] = wmma_bf16(a0, b0, acc[0][0]);
    acc[0][1] = wmma_bf16(a0, b1, acc[0][1]);
    acc[1][0] = wmma_bf16(a1, b0, acc[1][0]);
    acc[1][1] = wmma_bf16(a1, b1, acc[1][1]);
  }

  // C/D layout: lane l: n = l&15; VGPR v: m_local = v + (l>>4)*8
  const int n  = lane & 15;
  const int hi = lane >> 4;
  TO* Cb = C + (size_t)b * sC;
#pragma unroll
  for (int i = 0; i < 2; ++i) {
    const int m = tm * 32 + i * 16 + hi * 8;
    float rs = 1.0f;
    if (EPI == 2) rs = __builtin_amdgcn_rcpf(rowScale[(size_t)b * sScale + m]);
#pragma unroll
    for (int j = 0; j < 2; ++j) {
      const int nn = tn * 32 + j * 16 + n;
#pragma unroll
      for (int v = 0; v < 8; ++v) {
        float val = acc[i][j][v];
        if (HAS_BIAS) val += bias[nn];
        if (EPI == 1) val = __builtin_amdgcn_rcpf(1.0f + __expf(-val));
        if (EPI == 2) val *= rs;
        Cb[(size_t)(m + v) * ldc + nn] = (TO)val;
      }
    }
  }
}

// ---------------- cross GEMM: cross[b] = x[b] @ mu[b]^T --------------------
// mu[b] (64x256 bf16, 32KB) is staged into LDS once per block with CDNA5
// async global->LDS copies; 8 waves/block each compute a 32(M) x 64(K) strip
// (2x4 WMMA tiles, 8 WMMAs per k-step) streaming x fragments from global/L2.
__global__ void __launch_bounds__(256)
cross_gemm_kernel(const __bf16* __restrict__ x,    // (B, M, L)
                  const __bf16* __restrict__ mu,   // (B, K, L)
                  float* __restrict__ cross) {     // (B, M, K)
  __shared__ __bf16 smu[cK * cL];                  // 32 KB
  const int b = blockIdx.y;
  const int wave = threadIdx.x >> 5, lane = threadIdx.x & 31;

  // Stage mu[b] -> LDS: 2048 x 16B chunks, 256 threads, 8 rounds.
  {
    const char* gsrc = (const char*)(mu + (size_t)b * cK * cL);
    const unsigned lbase = (unsigned)(size_t)(&smu[0]);  // LDS offset in [31:0]
    for (int i = threadIdx.x; i < (cK * cL * 2) / 16; i += 256) {
      const unsigned lofs = lbase + i * 16;
      const char* g = gsrc + (size_t)i * 16;
      asm volatile("global_load_async_to_lds_b128 %0, %1, off"
                   :: "v"(lofs), "v"(g) : "memory");
    }
    wait_asynccnt0();
    __syncthreads();
  }

  const int m0 = (blockIdx.x * 8 + wave) * 32;
  const __bf16* Ab = x + ((size_t)b * cM + m0) * cL;

  v8f acc[2][4] = {};
  for (int kk = 0; kk < cL; kk += 32) {
    v16bf a0 = load_frag_a(Ab,           cL, lane, kk);
    v16bf a1 = load_frag_a(Ab + 16 * cL, cL, lane, kk);
    v16bf bf[4];
#pragma unroll
    for (int t = 0; t < 4; ++t)
      bf[t] = load_frag_b(&smu[(size_t)t * 16 * cL], cL, lane, kk);
#pragma unroll
    for (int t = 0; t < 4; ++t) {
      acc[0][t] = wmma_bf16(a0, bf[t], acc[0][t]);
      acc[1][t] = wmma_bf16(a1, bf[t], acc[1][t]);
    }
  }

  const int n = lane & 15, hi = lane >> 4;
  float* Cb = cross + ((size_t)b * cM + m0) * cK;
#pragma unroll
  for (int i = 0; i < 2; ++i)
#pragma unroll
    for (int t = 0; t < 4; ++t)
#pragma unroll
      for (int v = 0; v < 8; ++v) {
        const int m = i * 16 + v + hi * 8;
        Cb[(size_t)m * cK + t * 16 + n] = acc[i][t][v];
      }
}

// ---------------- weight transpose + f32->bf16 -----------------------------
__global__ void transpose_to_bf16_kernel(const float* __restrict__ src,
                                         __bf16* __restrict__ dst, int K, int N) {
  int idx = blockIdx.x * blockDim.x + threadIdx.x;
  if (idx >= K * N) return;
  int n = idx / K, k = idx % K;
  dst[idx] = (__bf16)src[(size_t)k * N + n];
}

// ---------------- z/x generation: one wave per (b,m) -----------------------
__global__ void zgen_kernel(const float* __restrict__ qp,
                            const float* __restrict__ eps,
                            __bf16* __restrict__ xbf, __bf16* __restrict__ xT,
                            float* __restrict__ x_sq) {
  const int wave = threadIdx.x >> 5, lane = threadIdx.x & 31;
  const int gm = blockIdx.x * (blockDim.x >> 5) + wave;
  if (gm >= cB * cM) return;
  const int b = gm / cM, m = gm % cM;
  const int n = m / cMC, mc = m % cMC;
  const float* qprow  = qp  + ((size_t)b * cN + n) * (2 * cL);
  const float* epsrow = eps + (((size_t)(b * cN + n)) * cMC + mc) * cL;
  __bf16* xrow = xbf + (size_t)gm * cL;
  __bf16* xTb  = xT  + (size_t)b * cL * cM;
  float ss = 0.f;
  for (int l = lane; l < cL; l += 32) {
    float xv = qprow[l] + epsrow[l] * __expf(0.5f * qprow[cL + l]);
    xrow[l] = (__bf16)xv;
    xTb[(size_t)l * cM + m] = (__bf16)xv;
    ss += xv * xv;
  }
  ss = wave_sum(ss);
  if (lane == 0) x_sq[gm] = ss;
}

// ---------------- mu/pi init ----------------------------------------------
__global__ void mu_init_kernel(const __bf16* __restrict__ xbf,
                               const int* __restrict__ init_idx,
                               __bf16* __restrict__ mu, float* __restrict__ pi) {
  int idx = blockIdx.x * blockDim.x + threadIdx.x;
  if (idx >= cB * cK * cL) return;
  int l = idx % cL, bk = idx / cL;
  int b = bk / cK, k = bk % cK;
  int src = init_idx[b * cK + k];
  mu[idx] = xbf[((size_t)b * cM + src) * cL + l];
  if (l == 0) pi[bk] = 1.0f / (float)cK;
}

// ---------------- mu_sq: one wave per (b,k) --------------------------------
__global__ void musq_kernel(const __bf16* __restrict__ mu,
                            float* __restrict__ mu_sq) {
  int wave = threadIdx.x >> 5, lane = threadIdx.x & 31;
  int bk = blockIdx.x * (blockDim.x >> 5) + wave;
  if (bk >= cB * cK) return;
  const __bf16* row = mu + (size_t)bk * cL;
  float ss = 0.f;
  for (int l = lane; l < cL; l += 32) { float v = (float)row[l]; ss += v * v; }
  ss = wave_sum(ss);
  if (lane == 0) mu_sq[bk] = ss;
}

// ---------------- softmax over K=64: one wave per (b,m) --------------------
__global__ void softmax_kernel(const float* __restrict__ cross,
                               const float* __restrict__ x_sq,
                               const float* __restrict__ mu_sq,
                               const float* __restrict__ pi,
                               __bf16* __restrict__ postT) {
  int wave = threadIdx.x >> 5, lane = threadIdx.x & 31;
  int gm = blockIdx.x * (blockDim.x >> 5) + wave;
  if (gm >= cB * cM) return;
  int b = gm / cM, m = gm % cM;
  const float* crow = cross + (size_t)gm * cK;
  float xs = x_sq[gm];
  int k0 = lane, k1 = lane + 32;
  float ll0 = -0.5f * (xs - 2.f * crow[k0] + mu_sq[b * cK + k0]) +
              (float)cL * LOG_NORM + __logf(pi[b * cK + k0]);
  float ll1 = -0.5f * (xs - 2.f * crow[k1] + mu_sq[b * cK + k1]) +
              (float)cL * LOG_NORM + __logf(pi[b * cK + k1]);
  float mx = wave_max(fmaxf(ll0, ll1));
  float e0 = __expf(ll0 - mx), e1 = __expf(ll1 - mx);
  float inv = __builtin_amdgcn_rcpf(wave_sum(e0 + e1));
  __bf16* pT = postT + (size_t)b * cK * cM;
  pT[(size_t)k0 * cM + m] = (__bf16)(e0 * inv);
  pT[(size_t)k1 * cM + m] = (__bf16)(e1 * inv);
}

// ---------------- Nk and pi update: one wave per (b,k) ---------------------
__global__ void nk_kernel(const __bf16* __restrict__ postT,
                          float* __restrict__ Nk, float* __restrict__ pi) {
  int wave = threadIdx.x >> 5, lane = threadIdx.x & 31;
  int bk = blockIdx.x * (blockDim.x >> 5) + wave;
  if (bk >= cB * cK) return;
  const __bf16* row = postT + (size_t)bk * cM;
  float s = 0.f;
  for (int m = lane; m < cM; m += 32) s += (float)row[m];
  s = wave_sum(s);
  if (lane == 0) {
    Nk[bk] = s;
    pi[bk] = s / (float)cM;  // sum_k Nk == M exactly
  }
}

// ---------------- final KL: one wave per (b,m), block-reduced atomic -------
__global__ void final_kernel(const float* __restrict__ cross,
                             const float* __restrict__ x_sq,
                             const float* __restrict__ mu_sq,
                             const float* __restrict__ pi,
                             const float* __restrict__ qp,
                             const float* __restrict__ eps,
                             float* __restrict__ accum) {
  __shared__ float wsum[8];
  int wave = threadIdx.x >> 5, lane = threadIdx.x & 31;
  int gm = blockIdx.x * (blockDim.x >> 5) + wave;
  float contrib = 0.f;
  if (gm < cB * cM) {
    int b = gm / cM, m = gm % cM;
    int n = m / cMC, mc = m % cMC;
    const float* qprow  = qp  + ((size_t)b * cN + n) * (2 * cL);
    const float* epsrow = eps + (((size_t)(b * cN + n)) * cMC + mc) * cL;
    float qz = 0.f;
    for (int l = lane; l < cL; l += 32) {
      float e = epsrow[l];
      qz += -0.5f * (qprow[cL + l] + e * e) + LOG_NORM;
    }
    qz = wave_sum(qz);
    const float* crow = cross + (size_t)gm * cK;
    float xs = x_sq[gm];
    int k0 = lane, k1 = lane + 32;
    float ll0 = -0.5f * (xs - 2.f * crow[k0] + mu_sq[b * cK + k0]) +
                (float)cL * LOG_NORM + __logf(pi[b * cK + k0]);
    float ll1 = -0.5f * (xs - 2.f * crow[k1] + mu_sq[b * cK + k1]) +
                (float)cL * LOG_NORM + __logf(pi[b * cK + k1]);
    float mx = wave_max(fmaxf(ll0, ll1));
    float s = wave_sum(__expf(ll0 - mx) + __expf(ll1 - mx));
    contrib = qz - (mx + __logf(s));
  }
  if (lane == 0) wsum[wave] = contrib;
  __syncthreads();
  if (threadIdx.x == 0) {
    float t = 0.f;
    for (int i = 0; i < 8; ++i) t += wsum[i];
    atomicAdd(accum, t);
  }
}

__global__ void init_acc_kernel(float* a) { if (threadIdx.x == 0) a[0] = 0.f; }
__global__ void finalize_kernel(const float* a, float* out) {
  out[0] = a[0] / (float)(cB * cN * cMC);
}

// ---------------------------------------------------------------------------
extern "C" void kernel_launch(void* const* d_in, const int* in_sizes, int n_in,
                              void* d_out, int out_size, void* d_ws,
                              size_t ws_size, hipStream_t stream) {
  const float* H        = (const float*)d_in[0];
  const float* W1       = (const float*)d_in[1];
  const float* b1       = (const float*)d_in[2];
  const float* W2       = (const float*)d_in[3];
  const float* b2       = (const float*)d_in[4];
  const float* eps      = (const float*)d_in[5];
  const int*   init_idx = (const int*)d_in[6];
  float* out = (float*)d_out;

  char* ws = (char*)d_ws;
  auto carve = [&](size_t bytes) -> char* {
    char* p = ws;
    ws += (bytes + 255) & ~(size_t)255;
    return p;
  };
  __bf16* W1t   = (__bf16*)carve((size_t)cHID * cHID * 2);
  __bf16* W2t   = (__bf16*)carve((size_t)2 * cL * cHID * 2);
  __bf16* h_bf  = (__bf16*)carve((size_t)cBN * cHID * 2);
  float*  qp    = (float*) carve((size_t)cBN * 2 * cL * 4);
  __bf16* xbf   = (__bf16*)carve((size_t)cB * cM * cL * 2);
  __bf16* xT    = (__bf16*)carve((size_t)cB * cM * cL * 2);
  float*  x_sq  = (float*) carve((size_t)cB * cM * 4);
  float*  cross = (float*) carve((size_t)cB * cM * cK * 4);
  __bf16* postT = (__bf16*)carve((size_t)cB * cK * cM * 2);
  __bf16* mu    = (__bf16*)carve((size_t)cB * cK * cL * 2);
  float*  mu_sq = (float*) carve((size_t)cB * cK * 4);
  float*  Nk    = (float*) carve((size_t)cB * cK * 4);
  float*  pi    = (float*) carve((size_t)cB * cK * 4);
  float*  acc   = (float*) carve(256);

  // 1. transpose/convert weights to bf16 NxK
  transpose_to_bf16_kernel<<<(cHID * cHID + 255) / 256, 256, 0, stream>>>(
      W1, W1t, cHID, cHID);
  transpose_to_bf16_kernel<<<(cHID * 2 * cL + 255) / 256, 256, 0, stream>>>(
      W2, W2t, cHID, 2 * cL);

  // 2. h = sigmoid(H @ W1 + b1)  -> bf16
  {
    int M = cBN, N = cHID, K = cHID;
    dim3 grid(((M / 32) * (N / 32) + 3) / 4, 1);
    wmma_gemm_kernel<float, __bf16, 1, true><<<grid, 128, 0, stream>>>(
        H, K, 0, W1t, K, 0, h_bf, N, 0, b1, nullptr, 0, M, N, K);
  }
  // 3. qp = h @ W2 + b2  -> f32
  {
    int M = cBN, N = 2 * cL, K = cHID;
    dim3 grid(((M / 32) * (N / 32) + 3) / 4, 1);
    wmma_gemm_kernel<__bf16, float, 0, true><<<grid, 128, 0, stream>>>(
        h_bf, K, 0, W2t, K, 0, qp, N, 0, b2, nullptr, 0, M, N, K);
  }

  // 4. x = q_mean + eps*exp(0.5*q_logvar); xbf, xT, x_sq
  zgen_kernel<<<(cB * cM) / 8, 256, 0, stream>>>(qp, eps, xbf, xT, x_sq);

  // 5. EM init
  mu_init_kernel<<<(cB * cK * cL + 255) / 256, 256, 0, stream>>>(
      xbf, init_idx, mu, pi);

  // 6. EM iterations
  for (int it = 0; it < EM_ITERS; ++it) {
    musq_kernel<<<(cB * cK + 7) / 8, 256, 0, stream>>>(mu, mu_sq);
    cross_gemm_kernel<<<dim3(cM / 256, cB), 256, 0, stream>>>(xbf, mu, cross);
    softmax_kernel<<<(cB * cM) / 8, 256, 0, stream>>>(cross, x_sq, mu_sq, pi,
                                                      postT);
    nk_kernel<<<(cB * cK + 7) / 8, 256, 0, stream>>>(postT, Nk, pi);
    {  // mu[b] = (postT[b] @ xT[b]^T) / Nk[b][k]
      dim3 grid(((cK / 32) * (cL / 32) + 3) / 4, cB);
      wmma_gemm_kernel<__bf16, __bf16, 2, false><<<grid, 128, 0, stream>>>(
          postT, cM, (size_t)cK * cM, xT, cM, (size_t)cL * cM,
          mu, cL, (size_t)cK * cL, nullptr, Nk, cK, cK, cL, cM);
    }
  }

  // 7. final log p(z) pieces with converged pi, mu
  musq_kernel<<<(cB * cK + 7) / 8, 256, 0, stream>>>(mu, mu_sq);
  cross_gemm_kernel<<<dim3(cM / 256, cB), 256, 0, stream>>>(xbf, mu, cross);
  init_acc_kernel<<<1, 32, 0, stream>>>(acc);
  final_kernel<<<(cB * cM) / 8, 256, 0, stream>>>(cross, x_sq, mu_sq, pi, qp,
                                                  eps, acc);
  finalize_kernel<<<1, 1, 0, stream>>>(acc, out);
}
